// EdgeConvLayer_2731599200751
// MI455X (gfx1250) — matrix operand order repaired and checked by reading, not compiled, not run on hardware
//
#include <hip/hip_runtime.h>

#define E_TOTAL   800000
#define N_NODES_C 50000
#define OUT_DIM_C 64
#define K_DIM     128   // 2 * IN_DIM

typedef __attribute__((ext_vector_type(16))) __bf16 v16bf;
typedef __attribute__((ext_vector_type(2)))  __bf16 v2bf;
typedef __attribute__((ext_vector_type(8)))  float  v8f;

// packed f32x2 -> bf16x2 conversion (native v_cvt_pk_bf16_f32 when available)
__device__ __forceinline__ v2bf f2bf2(float lo, float hi) {
#if __has_builtin(__builtin_amdgcn_cvt_pk_bf16_f32)
  return __builtin_amdgcn_cvt_pk_bf16_f32(lo, hi);
#else
  v2bf r; r[0] = (__bf16)lo; r[1] = (__bf16)hi; return r;
#endif
}

union AV { v16bf v; v2bf p[8]; };

__device__ __forceinline__ void load8(const float* __restrict__ p, float o[8]) {
  float4 x = *reinterpret_cast<const float4*>(p);
  float4 y = *reinterpret_cast<const float4*>(p + 4);
  o[0]=x.x; o[1]=x.y; o[2]=x.z; o[3]=x.w;
  o[4]=y.x; o[5]=y.y; o[6]=y.z; o[7]=y.w;
}

// ---- zero the accumulators (out, deg, stats) -------------------------------
__global__ void k_zero(float* a, int na, float* b, int nb, float* c, int nc) {
  int total = na + nb + nc;
  for (int i = blockIdx.x * blockDim.x + threadIdx.x; i < total;
       i += gridDim.x * blockDim.x) {
    if (i < na)            a[i] = 0.f;
    else if (i < na + nb)  b[i - na] = 0.f;
    else                   c[i - na - nb] = 0.f;
  }
}

// ---- W fp32 -> bf16 (B-matrix operand), row-major [64 x 128] ---------------
__global__ void k_convw(const float* __restrict__ W, unsigned short* __restrict__ Wb) {
  int i = 2 * (blockIdx.x * blockDim.x + threadIdx.x);
  if (i < OUT_DIM_C * K_DIM) {
    v2bf p = f2bf2(W[i], W[i + 1]);
    *reinterpret_cast<unsigned int*>(Wb + i) = __builtin_bit_cast(unsigned int, p);
  }
}

// ---- fold BN stats + bias + affine into per-channel scale/shift ------------
__global__ void k_stats(const float* __restrict__ gsum, const float* __restrict__ gsqr,
                        const float* __restrict__ b, const float* __restrict__ gamma,
                        const float* __restrict__ beta,
                        float* __restrict__ scale, float* __restrict__ shift) {
  int c = threadIdx.x;
  if (c < OUT_DIM_C) {
    const float inv_e = 1.0f / (float)E_TOTAL;
    float mu  = gsum[c] * inv_e;              // stats include bias (added in pass 1)
    float var = gsqr[c] * inv_e - mu * mu;
    float is  = rsqrtf(var + 1e-5f);
    float sc  = gamma[c] * is;
    scale[c]  = sc;
    shift[c]  = (b[c] - mu) * sc + beta[c];   // y = acc*sc + shift  (acc has no bias)
  }
}

// ---- main tile kernel: one wave32 per 16-edge tile, 16x16x32 bf16 WMMA -----
// PASS 1: accumulate per-channel sum / sumsq of h = m@W^T + b
// PASS 2: recompute h, apply relu(h*scale+shift), scatter-atomic into out/deg
template <int PASS>
__global__ __launch_bounds__(256) void k_edge(
    const float* __restrict__ feat,
    const long long* __restrict__ srcIdx,
    const long long* __restrict__ dstIdx,
    const unsigned short* __restrict__ Wb,
    const float* __restrict__ bias,
    const float* __restrict__ scale,
    const float* __restrict__ shift,
    float* __restrict__ gsum, float* __restrict__ gsqr,
    float* __restrict__ out, float* __restrict__ deg)
{
  __shared__ __align__(32) unsigned short shW[OUT_DIM_C * K_DIM];  // 16 KB bf16 W
  __shared__ float sSum[64], sSqr[64];
  __shared__ int   sDst[8][16];

  const int tid = threadIdx.x;

  // cooperatively stage bf16 W into LDS (uint4 = 16B chunks)
  {
    const uint4* g = reinterpret_cast<const uint4*>(Wb);
    uint4*       s = reinterpret_cast<uint4*>(shW);
    for (int i = tid; i < (OUT_DIM_C * K_DIM) / 8; i += 256) s[i] = g[i];
  }
  if (PASS == 1 && tid < 128) {
    if (tid < 64) sSum[tid] = 0.f; else sSqr[tid - 64] = 0.f;
  }

  const int lane  = tid & 31;
  const int wave  = tid >> 5;
  const int row   = lane & 15;   // A-matrix row (edge within tile), C-matrix column
  const int khalf = lane >> 4;   // which K half-group this lane holds

  const long long eb = ((long long)blockIdx.x * 8 + wave) * 16;  // tile base edge
  const long long e  = eb + row;
  const long long si = srcIdx[e];
  const long long di = dstIdx[e];
  if (PASS == 2 && lane < 16) sDst[wave][lane] = (int)di;
  __syncthreads();

  const float* fi = feat + di * 64;   // x_i = feature[dst]
  const float* fj = feat + si * 64;   // x_j = feature[src]

  // load this lane's 32 x_i floats once; reused by both halves of m
  // group g covers K-offsets g*16 + khalf*8 + [0,8)
  float fiR[4][8];
  #pragma unroll
  for (int g = 0; g < 4; ++g) load8(fi + g * 16 + khalf * 8, fiR[g]);

  const v8f vz = {0.f,0.f,0.f,0.f,0.f,0.f,0.f,0.f};
  v8f acc[4] = {vz, vz, vz, vz};      // 4 N-tiles of 16 channels

  #pragma unroll
  for (int kk = 0; kk < 4; ++kk) {    // K = 128 in chunks of 32
    const int kb = kk * 32;
    AV a;
    if (kk < 2) {                     // m[0:64] = x_i
      const float* gA = fiR[2 * kk];
      const float* gB = fiR[2 * kk + 1];
      #pragma unroll
      for (int i = 0; i < 4; ++i) {
        a.p[i]     = f2bf2(gA[2*i], gA[2*i+1]);
        a.p[i + 4] = f2bf2(gB[2*i], gB[2*i+1]);
      }
    } else {                          // m[64:128] = x_j - x_i
      float ja[8], jb[8];
      const int q = (kk - 2) * 32 + khalf * 8;
      load8(fj + q,      ja);
      load8(fj + q + 16, jb);
      const float* iA = fiR[2 * (kk - 2)];
      const float* iB = fiR[2 * (kk - 2) + 1];
      #pragma unroll
      for (int i = 0; i < 4; ++i) {
        a.p[i]     = f2bf2(ja[2*i] - iA[2*i], ja[2*i+1] - iA[2*i+1]);
        a.p[i + 4] = f2bf2(jb[2*i] - iB[2*i], jb[2*i+1] - iB[2*i+1]);
      }
    }

    #pragma unroll
    for (int nt = 0; nt < 4; ++nt) {
      const int n = nt * 16 + row;    // B column this lane holds
      union { v16bf v; uint4 u[2]; } bu;
      bu.u[0] = *reinterpret_cast<const uint4*>(shW + n * K_DIM + kb + khalf * 8);        // K kb+8h+[0,8), 16B aligned
      bu.u[1] = *reinterpret_cast<const uint4*>(shW + n * K_DIM + kb + 16 + khalf * 8);   // K kb+16+8h+[0,8)
      const v16bf bfrag = bu.v;
      acc[nt] = __builtin_amdgcn_wmma_f32_16x16x32_bf16(
          false, a.v, false, bfrag, (short)0, acc[nt], false, false);
    }
  }

  if (PASS == 1) {
    #pragma unroll
    for (int nt = 0; nt < 4; ++nt) {
      const int col = nt * 16 + row;
      const float bv = bias[col];
      float s1 = 0.f, s2 = 0.f;
      #pragma unroll
      for (int r = 0; r < 8; ++r) {
        float h = acc[nt][r] + bv;
        s1 += h; s2 += h * h;
      }
      atomicAdd(&sSum[col], s1);
      atomicAdd(&sSqr[col], s2);
    }
    __syncthreads();
    if (tid < 64)       atomicAdd(&gsum[tid], sSum[tid]);
    else if (tid < 128) atomicAdd(&gsqr[tid - 64], sSqr[tid - 64]);
  } else {
    __syncthreads();
    #pragma unroll
    for (int nt = 0; nt < 4; ++nt) {
      const int col = nt * 16 + row;
      const float sc = scale[col], sh = shift[col];
      #pragma unroll
      for (int r = 0; r < 8; ++r) {
        const int M = r + khalf * 8;              // C/D row -> edge within tile
        const float y = fmaxf(acc[nt][r] * sc + sh, 0.f);
        const int d = sDst[wave][M];
        atomicAdd(out + (long long)d * 64 + col, y);
      }
    }
    if (lane < 16) atomicAdd(&deg[sDst[wave][lane]], 1.0f);
  }
}

// ---- final mean: out /= max(deg, 1) ----------------------------------------
__global__ void k_div(float* __restrict__ out, const float* __restrict__ deg) {
  int i = blockIdx.x * blockDim.x + threadIdx.x;
  if (i < N_NODES_C * OUT_DIM_C)
    out[i] = out[i] / fmaxf(deg[i >> 6], 1.0f);
}

extern "C" void kernel_launch(void* const* d_in, const int* in_sizes, int n_in,
                              void* d_out, int out_size, void* d_ws, size_t ws_size,
                              hipStream_t stream) {
  const float*     feat  = (const float*)d_in[0];
  const long long* ei    = (const long long*)d_in[1];
  const float*     W     = (const float*)d_in[2];
  const float*     b     = (const float*)d_in[3];
  const float*     gamma = (const float*)d_in[4];
  const float*     beta  = (const float*)d_in[5];
  float* out = (float*)d_out;

  char* ws = (char*)d_ws;
  unsigned short* Wb    = (unsigned short*)ws;        // 16384 B  bf16 W
  float*          gsum  = (float*)(ws + 16384);       //   256 B
  float*          gsqr  = (float*)(ws + 16640);       //   256 B (contiguous w/ gsum)
  float*          scale = (float*)(ws + 16896);
  float*          shift = (float*)(ws + 17152);
  float*          deg   = (float*)(ws + 17408);       // 200000 B

  const long long* srcIdx = ei;             // edge_index[0] = src (x_j)
  const long long* dstIdx = ei + E_TOTAL;   // edge_index[1] = dst (x_i, aggregation)

  k_zero<<<2048, 256, 0, stream>>>(out, N_NODES_C * OUT_DIM_C, deg, N_NODES_C, gsum, 128);
  k_convw<<<(OUT_DIM_C * K_DIM / 2 + 255) / 256, 256, 0, stream>>>(W, Wb);
  k_edge<1><<<E_TOTAL / 16 / 8, 256, 0, stream>>>(feat, srcIdx, dstIdx, Wb, b,
                                                  scale, shift, gsum, gsqr, out, deg);
  k_stats<<<1, 64, 0, stream>>>(gsum, gsqr, b, gamma, beta, scale, shift);
  k_edge<2><<<E_TOTAL / 16 / 8, 256, 0, stream>>>(feat, srcIdx, dstIdx, Wb, b,
                                                  scale, shift, gsum, gsqr, out, deg);
  k_div<<<(N_NODES_C * OUT_DIM_C + 255) / 256, 256, 0, stream>>>(out, deg);
}
